// HeterogeneousCriterionGNN_21251498180622
// MI455X (gfx1250) — matrix operand order repaired and checked by reading, not compiled, not run on hardware
//
#include <hip/hip_runtime.h>
#include <hip/hip_bf16.h>

typedef _Float16 v16h __attribute__((ext_vector_type(16)));
typedef float    v8f  __attribute__((ext_vector_type(8)));
typedef unsigned int v4u __attribute__((ext_vector_type(4)));

union Frag { v16h h; v4u u[2]; };

#define BM 128
#define BN 128
#define BK 32

// ---------------------------------------------------------------------------
// Tiled WMMA GEMM: C[M,N] = A[M,K] * B[K,N] (+bias) (+optional ReLU).
// A,B,C fp32 in global; converted to f16 while staging; f32 accumulation via
// v_wmma_f32_16x16x32_f16. 256 threads = 8 waves; 128x128 block tile; each
// wave owns a 32x64 sub-tile (2x4 WMMA tiles -> 8 WMMAs per K-step).
// Register-prefetched staging + ping-pong LDS, one barrier per K-step.
// ---------------------------------------------------------------------------
__global__ __launch_bounds__(256) void gemm_wmma_f16(
    const float* __restrict__ A, const float* __restrict__ B,
    const float* __restrict__ bias, float* __restrict__ C,
    int M, int N, int K, int act)
{
    __shared__ _Float16 lA[2][BM][BK];   // 2 x 8KB
    __shared__ _Float16 lB[2][BN][BK];   // 2 x 8KB (B transposed: [n][k])

    const int tid  = threadIdx.x;
    const int wid  = tid >> 5;
    const int lane = tid & 31;
    const int m0 = blockIdx.x * BM;
    const int n0 = blockIdx.y * BN;
    const int wm = (wid >> 1) * 32;   // wave row offset (4 wave-rows)
    const int wn = (wid & 1) * 64;    // wave col offset (2 wave-cols x 64)

    v8f acc[2][4];
    const v8f vzero = {0.f,0.f,0.f,0.f,0.f,0.f,0.f,0.f};
#pragma unroll
    for (int i = 0; i < 2; ++i)
#pragma unroll
        for (int j = 0; j < 4; ++j) acc[i][j] = vzero;

    // staging assignments
    const int ar = tid >> 1;           // A row 0..127
    const int ac = (tid & 1) * 16;     // A col half 0/16
    const int bn = tid & 127;          // B col 0..127 (coalesced across lanes)
    const int bk = (tid >> 7) * 16;    // B k half 0/16

    // fragment addressing (ISA 7.12.2 layouts, wave32)
    const int kbA  = (lane >> 4) * 8;   // A: lanes16-31 start at K=8 (then +16)
    const int kbB  = (lane >> 4) * 8;   // B: lanes16-31 start at K=8 (then +16)
    const int nlo  = lane & 15;
    const int rsel = (lane >> 4) * 8;   // C/D: lanes16-31 -> rows +8

    const int gm = m0 + ar;
    const bool aok = gm < M;
    const int gn = n0 + bn;
    const bool bok = gn < N;

    float aReg[16];
    float bReg[16];

    auto loadRegs = [&](int k0) {
        if (aok) {
            const float* ap = A + (size_t)gm * K + k0 + ac;
#pragma unroll
            for (int i = 0; i < 16; i += 4) {
                float4 v = *(const float4*)(ap + i);
                aReg[i + 0] = v.x; aReg[i + 1] = v.y;
                aReg[i + 2] = v.z; aReg[i + 3] = v.w;
            }
            if (k0 + BK < K) __builtin_prefetch(ap + BK, 0, 1);
        } else {
#pragma unroll
            for (int i = 0; i < 16; ++i) aReg[i] = 0.f;
        }
        if (bok) {
            const float* bp = B + (size_t)(k0 + bk) * N + gn;
#pragma unroll
            for (int i = 0; i < 16; ++i) bReg[i] = bp[(size_t)i * N];
        } else {
#pragma unroll
            for (int i = 0; i < 16; ++i) bReg[i] = 0.f;
        }
    };

    loadRegs(0);
    int buf = 0;

    for (int k0 = 0; k0 < K; k0 += BK) {
        // ---- commit staged registers (f32 -> f16) into LDS buffer ----
#pragma unroll
        for (int i = 0; i < 16; ++i) lA[buf][ar][ac + i] = (_Float16)aReg[i];
#pragma unroll
        for (int i = 0; i < 16; ++i) lB[buf][bn][bk + i] = (_Float16)bReg[i];
        __syncthreads();

        // ---- issue next tile's global loads (overlap with WMMA) ----
        if (k0 + BK < K) loadRegs(k0 + BK);

        // ---- fragments + 8 WMMAs ----
        Frag af[2], bf[4];
#pragma unroll
        for (int mt = 0; mt < 2; ++mt) {
            const _Float16* p = &lA[buf][wm + mt * 16 + nlo][kbA];
            af[mt].u[0] = *(const v4u*)(p);
            af[mt].u[1] = *(const v4u*)(p + 16);
        }
#pragma unroll
        for (int nt = 0; nt < 4; ++nt) {
            const _Float16* p = &lB[buf][wn + nt * 16 + nlo][kbB];
            bf[nt].u[0] = *(const v4u*)(p);
            bf[nt].u[1] = *(const v4u*)(p + 16);
        }
#pragma unroll
        for (int mt = 0; mt < 2; ++mt)
#pragma unroll
            for (int nt = 0; nt < 4; ++nt)
                acc[mt][nt] = __builtin_amdgcn_wmma_f32_16x16x32_f16(
                    false, af[mt].h, false, bf[nt].h,
                    (short)0, acc[mt][nt], false, false);
        buf ^= 1;
    }

    // ---- epilogue: bias + activation + store ----
#pragma unroll
    for (int mt = 0; mt < 2; ++mt) {
#pragma unroll
        for (int nt = 0; nt < 4; ++nt) {
            const int col = n0 + wn + nt * 16 + nlo;
            const float bv = (bias && col < N) ? bias[col] : 0.f;
#pragma unroll
            for (int r = 0; r < 8; ++r) {
                const int row = m0 + wm + mt * 16 + rsel + r;
                if (row < M && col < N) {
                    float v = acc[mt][nt][r] + bv;
                    if (act == 1) v = v > 0.f ? v : 0.f;
                    C[(size_t)row * N + col] = v;
                }
            }
        }
    }
}

// ---------------------------------------------------------------------------
// Order-preserving uint encoding for float atomic max
// ---------------------------------------------------------------------------
__device__ __forceinline__ unsigned enc_f(float x) {
    unsigned u = __float_as_uint(x);
    return (u & 0x80000000u) ? ~u : (u | 0x80000000u);
}
__device__ __forceinline__ float dec_f(unsigned u) {
    return (u & 0x80000000u) ? __uint_as_float(u & 0x7fffffffu)
                             : __uint_as_float(~u);
}
#define ENC_NEGINF 0x007FFFFFu

__global__ void init_mz_k(unsigned* __restrict__ m, float* __restrict__ z, int n) {
    int i = blockIdx.x * blockDim.x + threadIdx.x;
    if (i >= n) return;
    m[i] = ENC_NEGINF;
    z[i] = 0.f;
}

__global__ void init_out_k(float* __restrict__ out, const float* __restrict__ b0,
                           const float* __restrict__ b1, size_t total) {
    size_t i = (size_t)blockIdx.x * blockDim.x + threadIdx.x;
    if (i >= total) return;
    int c = (int)(i & 255);
    out[i] = b0[c] + (b1 ? b1[c] : 0.f);
}

// e = leaky_relu(a_src[si] + a_dst[di]); atomic segment-max
__global__ void edge_pass1_k(const int* __restrict__ si, const int* __restrict__ di,
                             const float* __restrict__ asrc, const float* __restrict__ adst,
                             float* __restrict__ e, unsigned* __restrict__ menc, int E) {
    int i = blockIdx.x * blockDim.x + threadIdx.x;
    if (i >= E) return;
    int s = si[i], d = di[i];
#pragma unroll
    for (int h = 0; h < 4; ++h) {
        float v = asrc[s * 4 + h] + adst[d * 4 + h];
        v = v > 0.f ? v : 0.2f * v;
        e[i * 4 + h] = v;
        atomicMax(&menc[d * 4 + h], enc_f(v));
    }
}

// ex = exp(e - m[di]); atomic segment-sum into z; e overwritten with ex
__global__ void edge_pass2_k(const int* __restrict__ di, float* __restrict__ e,
                             const unsigned* __restrict__ menc, float* __restrict__ z, int E4) {
    int i = blockIdx.x * blockDim.x + threadIdx.x;
    if (i >= E4) return;
    int d = di[i >> 2];
    int h = i & 3;
    float m = dec_f(menc[d * 4 + h]);
    float ex = __expf(e[i] - m);
    e[i] = ex;
    atomicAdd(&z[d * 4 + h], ex);
}

// wave-per-edge: out[di] += (ex/z[di]) * h_src[si]  (256 channels, 4 heads)
__global__ void edge_pass3_k(const int* __restrict__ si, const int* __restrict__ di,
                             const float* __restrict__ e, const float* __restrict__ z,
                             const float* __restrict__ hsrc, float* __restrict__ out, int E) {
    int gw = blockIdx.x * 8 + (threadIdx.x >> 5);
    if (gw >= E) return;
    int lane = threadIdx.x & 31;
    int s = si[gw], d = di[gw];
    float al[4];
#pragma unroll
    for (int h = 0; h < 4; ++h) al[h] = e[gw * 4 + h] / z[d * 4 + h];
    const float* hp = hsrc + (size_t)s * 256;
    float* op = out + (size_t)d * 256;
#pragma unroll
    for (int j = 0; j < 8; ++j) {
        int c = lane + 32 * j;
        atomicAdd(&op[c], al[j >> 1] * hp[c]);
    }
}

__global__ void elu_k(const float* __restrict__ in, float* __restrict__ out, size_t n) {
    size_t i = (size_t)blockIdx.x * blockDim.x + threadIdx.x;
    if (i >= n) return;
    float x = in[i];
    out[i] = x > 0.f ? x : (__expf(x) - 1.f);
}

// wave per (node,head): a[n,h] = dot(h[n, h*64 : h*64+64], att[h])
__global__ void attvec_k(const float* __restrict__ h, const float* __restrict__ att,
                         float* __restrict__ a, int NH) {
    int gw = blockIdx.x * 8 + (threadIdx.x >> 5);
    if (gw >= NH) return;
    int lane = threadIdx.x & 31;
    int n = gw >> 2, hd = gw & 3;
    const float* hp = h + (size_t)n * 256 + hd * 64;
    const float* ap = att + hd * 64;
    float s = hp[lane] * ap[lane] + hp[lane + 32] * ap[lane + 32];
#pragma unroll
    for (int off = 16; off; off >>= 1) s += __shfl_xor(s, off, 32);
    if (lane == 0) a[gw] = s;
}

// wave per node: out[n] = dot(t[n,:128], W2) + b2
__global__ void head2_k(const float* __restrict__ t, const float* __restrict__ W2,
                        const float* __restrict__ b2, float* __restrict__ out, int N) {
    int gw = blockIdx.x * 8 + (threadIdx.x >> 5);
    if (gw >= N) return;
    int lane = threadIdx.x & 31;
    const float* tp = t + (size_t)gw * 128;
    float s = tp[lane] * W2[lane] + tp[lane + 32] * W2[lane + 32]
            + tp[lane + 64] * W2[lane + 64] + tp[lane + 96] * W2[lane + 96];
#pragma unroll
    for (int off = 16; off; off >>= 1) s += __shfl_xor(s, off, 32);
    if (lane == 0) out[gw] = s + b2[0];
}

// ---------------------------------------------------------------------------
extern "C" void kernel_launch(void* const* d_in, const int* in_sizes, int n_in,
                              void* d_out, int out_size, void* d_ws, size_t ws_size,
                              hipStream_t stream) {
    const float* x_c  = (const float*)d_in[0];
    const float* x_s  = (const float*)d_in[1];
    const int* eq_s   = (const int*)d_in[2];
    const int* eq_d   = (const int*)d_in[3];
    const int* erq_s  = (const int*)d_in[4];
    const int* erq_d  = (const int*)d_in[5];
    const int* es_s   = (const int*)d_in[6];
    const int* es_d   = (const int*)d_in[7];

    const int NC = in_sizes[0] / 768;
    const int NS = in_sizes[1] / 768;
    const int Eq = in_sizes[2], Erq = in_sizes[4], Es = in_sizes[6];
    int Emax = Eq > Erq ? Eq : Erq; if (Es > Emax) Emax = Es;

    const float* P[64];
    for (int i = 0; i < n_in && i < 64; ++i) P[i] = (const float*)d_in[i];

    struct Gat { const float *W, *as, *ad, *b; } L[2][3];
    const float *critW, *critB, *sentW, *sentB;
    const float *chW1, *chb1, *chW2, *chb2, *shW1, *shb1, *shW2, *shb2;

    if (in_sizes[8] == 768 * 256) {
        // dict insertion order flattening
        critW = P[8]; critB = P[9]; sentW = P[10]; sentB = P[11];
        int k = 12;
        for (int l = 0; l < 2; ++l)
            for (int et = 0; et < 3; ++et) {          // q, rq, sim
                L[l][et].W = P[k]; L[l][et].as = P[k + 1];
                L[l][et].ad = P[k + 2]; L[l][et].b = P[k + 3];
                k += 4;
            }
        chW1 = P[36]; chb1 = P[37]; chW2 = P[38]; chb2 = P[39];
        shW1 = P[40]; shb1 = P[41]; shW2 = P[42]; shb2 = P[43];
    } else {
        // jax tree_leaves (sorted dict keys) flattening
        chW1 = P[8]; chW2 = P[9]; chb1 = P[10]; chb2 = P[11];
        critW = P[12]; critB = P[13];
        int k = 14;
        for (int l = 0; l < 2; ++l)
            for (int et = 0; et < 3; ++et) {          // q, rq, sim ; W, att_dst, att_src, b
                L[l][et].W = P[k]; L[l][et].ad = P[k + 1];
                L[l][et].as = P[k + 2]; L[l][et].b = P[k + 3];
                k += 4;
            }
        shW1 = P[38]; shW2 = P[39]; shb1 = P[40]; shb2 = P[41];
        sentW = P[42]; sentB = P[43];
    }

    // ---- workspace layout ----
    char* base = (char*)d_ws;
    size_t off = 0;
    auto alc = [&](size_t nfloats) -> float* {
        float* p = (float*)(base + off);
        off += (nfloats * sizeof(float) + 255) & ~(size_t)255;
        return p;
    };
    float* xc   = alc((size_t)NC * 256);
    float* xs   = alc((size_t)NS * 256);
    float* hA   = alc((size_t)NS * 256);  // h_q_dst, recycled as out_s
    float* hB   = alc((size_t)NS * 256);  // h_sim, recycled as head tmp
    float* hC   = alc((size_t)NS * 256);  // h_rq_src
    float* hQs  = alc((size_t)NC * 256);  // h_q_src
    float* hRd  = alc((size_t)NC * 256);  // h_rq_dst
    float* outC = alc((size_t)NC * 256);
    float* aQs  = alc((size_t)NC * 4);
    float* aQd  = alc((size_t)NS * 4);
    float* aSs  = alc((size_t)NS * 4);
    float* aSd  = alc((size_t)NS * 4);
    float* aRs  = alc((size_t)NS * 4);
    float* aRd  = alc((size_t)NC * 4);
    unsigned* mQ = (unsigned*)alc((size_t)NS * 4);
    float*    zQ = alc((size_t)NS * 4);
    unsigned* mS = (unsigned*)alc((size_t)NS * 4);
    float*    zS = alc((size_t)NS * 4);
    unsigned* mR = (unsigned*)alc((size_t)NC * 4);
    float*    zR = alc((size_t)NC * 4);
    float* eBuf = alc((size_t)Emax * 4);

    auto GEMM = [&](const float* A, const float* Bm, const float* bias, float* Cm,
                    int M, int N, int K, int act) {
        dim3 g((unsigned)((M + BM - 1) / BM), (unsigned)((N + BN - 1) / BN));
        gemm_wmma_f16<<<g, 256, 0, stream>>>(A, Bm, bias, Cm, M, N, K, act);
    };
    auto ATTVEC = [&](const float* h, const float* att, float* a, int NH) {
        attvec_k<<<(NH + 7) / 8, 256, 0, stream>>>(h, att, a, NH);
    };

    // ---- input projections ----
    GEMM(x_c, critW, critB, xc, NC, 256, 768, 0);
    GEMM(x_s, sentW, sentB, xs, NS, 256, 768, 0);

    for (int l = 0; l < 2; ++l) {
        const Gat& q = L[l][0];
        const Gat& rq = L[l][1];
        const Gat& sim = L[l][2];

        // linear transforms (no bias inside GAT lin)
        GEMM(xc, q.W,   nullptr, hQs, NC, 256, 256, 0);
        GEMM(xs, q.W,   nullptr, hA,  NS, 256, 256, 0);
        GEMM(xs, sim.W, nullptr, hB,  NS, 256, 256, 0);
        GEMM(xs, rq.W,  nullptr, hC,  NS, 256, 256, 0);
        GEMM(xc, rq.W,  nullptr, hRd, NC, 256, 256, 0);

        // attention coefficients
        ATTVEC(hQs, q.as,  aQs, NC * 4);
        ATTVEC(hA,  q.ad,  aQd, NS * 4);
        ATTVEC(hB,  sim.as, aSs, NS * 4);
        ATTVEC(hB,  sim.ad, aSd, NS * 4);
        ATTVEC(hC,  rq.as, aRs, NS * 4);
        ATTVEC(hRd, rq.ad, aRd, NC * 4);

        // init softmax state + output accumulators (with GAT biases)
        init_mz_k<<<(NS * 4 + 255) / 256, 256, 0, stream>>>(mQ, zQ, NS * 4);
        init_mz_k<<<(NS * 4 + 255) / 256, 256, 0, stream>>>(mS, zS, NS * 4);
        init_mz_k<<<(NC * 4 + 255) / 256, 256, 0, stream>>>(mR, zR, NC * 4);
        float* outS = hA;   // h_q_dst no longer needed
        {
            size_t tS = (size_t)NS * 256, tC = (size_t)NC * 256;
            init_out_k<<<(unsigned)((tS + 255) / 256), 256, 0, stream>>>(outS, q.b, sim.b, tS);
            init_out_k<<<(unsigned)((tC + 255) / 256), 256, 0, stream>>>(outC, rq.b, nullptr, tC);
        }

        // conv: crit -> sent (queries)
        edge_pass1_k<<<(Eq + 255) / 256, 256, 0, stream>>>(eq_s, eq_d, aQs, aQd, eBuf, mQ, Eq);
        edge_pass2_k<<<(Eq * 4 + 255) / 256, 256, 0, stream>>>(eq_d, eBuf, mQ, zQ, Eq * 4);
        edge_pass3_k<<<(Eq + 7) / 8, 256, 0, stream>>>(eq_s, eq_d, eBuf, zQ, hQs, outS, Eq);
        // conv: sent -> sent (similar)
        edge_pass1_k<<<(Es + 255) / 256, 256, 0, stream>>>(es_s, es_d, aSs, aSd, eBuf, mS, Es);
        edge_pass2_k<<<(Es * 4 + 255) / 256, 256, 0, stream>>>(es_d, eBuf, mS, zS, Es * 4);
        edge_pass3_k<<<(Es + 7) / 8, 256, 0, stream>>>(es_s, es_d, eBuf, zS, hB, outS, Es);
        // conv: sent -> crit (rev_queries)
        edge_pass1_k<<<(Erq + 255) / 256, 256, 0, stream>>>(erq_s, erq_d, aRs, aRd, eBuf, mR, Erq);
        edge_pass2_k<<<(Erq * 4 + 255) / 256, 256, 0, stream>>>(erq_d, eBuf, mR, zR, Erq * 4);
        edge_pass3_k<<<(Erq + 7) / 8, 256, 0, stream>>>(erq_s, erq_d, eBuf, zR, hC, outC, Erq);

        // activations -> next layer inputs
        {
            size_t tS = (size_t)NS * 256, tC = (size_t)NC * 256;
            elu_k<<<(unsigned)((tS + 255) / 256), 256, 0, stream>>>(outS, xs, tS);
            elu_k<<<(unsigned)((tC + 255) / 256), 256, 0, stream>>>(outC, xc, tC);
        }
    }

    // ---- heads: relu(x @ W1 + b1) @ W2 + b2 ----
    float* tmp = hB;
    float* out = (float*)d_out;
    GEMM(xc, chW1, chb1, tmp, NC, 128, 256, 1);
    head2_k<<<(NC + 7) / 8, 256, 0, stream>>>(tmp, chW2, chb2, out, NC);
    GEMM(xs, shW1, shb1, tmp, NS, 128, 256, 1);
    head2_k<<<(NS + 7) / 8, 256, 0, stream>>>(tmp, shW2, shb2, out + NC, NS);
}
